// SelfAttentionLayer_27376121545426
// MI455X (gfx1250) — matrix operand order repaired and checked by reading, not compiled, not run on hardware
//
#include <hip/hip_runtime.h>
#include <hip/hip_bf16.h>

// CDNA5 / gfx1250 fused single-query attention (decode attention), flash-decoding style.
//
// Algebraic reduction:
//   scores[b,n] = x[b,n,:] . qt[b],  qt[b] = Wk^T @ (Wq @ query[b])
//   out[b]      = Wv @ ( sum_n softmax(scores)[n] * x[b,n,:] ) = Wv @ (c[b] / l[b])
// => one streaming pass over x (512MB) at HBM roofline (~22us), WMMA for the
//    per-tile score matvec and the weighted row accumulation.

typedef __attribute__((ext_vector_type(16))) __bf16 v16bf;
typedef __attribute__((ext_vector_type(8)))  __bf16 v8bf;
typedef __attribute__((ext_vector_type(8)))  float  v8f;

#define B_    128
#define N_    8192
#define D_    128
#define CH    16                 // chunks per batch (flash-decoding split)
#define CHUNK (N_ / CH)          // 512 rows per workgroup
#define WAVES 4
#define RPW   (CHUNK / WAVES)    // 128 rows per wave
#define TILE  32                 // rows per inner iteration
#define ITERS (RPW / TILE)       // 4
#define NEGV  (-1e9f)
#define FNEG  (-3.0e38f)

__global__ __launch_bounds__(128) void attn_partial_kernel(
    const float* __restrict__ query, const float* __restrict__ x,
    const int* __restrict__ amask, const float* __restrict__ Wq,
    const float* __restrict__ Wk, float* __restrict__ ws)
{
    __shared__ float  pq[D_];
    __shared__ float  qt[D_];
    __shared__ __bf16 xb[WAVES][TILE * D_];   // 32KB: per-wave bf16 x tile
    __shared__ float  madd[WAVES][TILE];
    __shared__ float  pc[WAVES][D_];
    __shared__ float  pm[WAVES];
    __shared__ float  pl[WAVES];

    const int tid  = threadIdx.x;
    const int wv   = tid >> 5;
    const int lane = tid & 31;
    const int b    = blockIdx.x / CH;
    const int ch   = blockIdx.x % CH;

    // ---- per-batch setup: Pq = Wq @ query[b]; qt = Wk^T @ Pq ----
    {
        const float* qv = query + (size_t)b * D_;
        const float* wr = Wq + (size_t)tid * D_;
        float s = 0.f;
        #pragma unroll 4
        for (int d = 0; d < D_; ++d) s += wr[d] * qv[d];
        pq[tid] = s;
    }
    __syncthreads();
    {
        float s = 0.f;
        #pragma unroll 4
        for (int e = 0; e < D_; ++e) s += pq[e] * Wk[(size_t)e * D_ + tid];
        qt[tid] = s;
    }
    __syncthreads();

    // qt as WMMA B operand (all columns): element s holds K = kk*32 + 16*(s/8) + 8*(lane/16) + s%8.
    v16bf qB[4];
    #pragma unroll
    for (int kk = 0; kk < 4; ++kk) {
        v16bf t = {};
        #pragma unroll
        for (int s = 0; s < 16; ++s)
            t[s] = (__bf16)qt[kk * 32 + ((s >> 3) << 4) + ((lane >> 4) << 3) + (s & 7)];
        qB[kk] = t;
    }

    float m = FNEG, l = 0.f;
    v8f cacc[8];
    #pragma unroll
    for (int t = 0; t < 8; ++t) cacc[t] = {};

    const int  row_base = ch * CHUNK + wv * RPW;
    const bool ok       = ((lane & 15) == 0);   // lanes 0 and 16 own score rows
    const int  sub_off  = (lane >= 16) ? 8 : 0;
    const int  kpart    = (lane >> 4) * 8;

    for (int it = 0; it < ITERS; ++it) {
        const int r0 = row_base + it * TILE;

        // ---- coalesced load of 32x128 fp32 tile -> bf16 in LDS ----
        {
            const float4* xt = (const float4*)(x + ((size_t)b * N_ + r0) * D_);
            __bf16* lb = &xb[wv][0];
            #pragma unroll
            for (int u = 0; u < 16; ++u) {
                const int g = u * 64 + lane * 2;    // float4 index within tile
                float4 a = xt[g];
                float4 c = xt[g + 1];
                v8bf p;
                p[0] = (__bf16)a.x; p[1] = (__bf16)a.y; p[2] = (__bf16)a.z; p[3] = (__bf16)a.w;
                p[4] = (__bf16)c.x; p[5] = (__bf16)c.y; p[6] = (__bf16)c.z; p[7] = (__bf16)c.w;
                *(v8bf*)(lb + (size_t)g * 4) = p;
            }
            int mk = amask[(size_t)b * N_ + r0 + lane];
            madd[wv][lane] = (mk == 0) ? 1.0f : 0.0f;
        }
        asm volatile("s_wait_dscnt 0" ::: "memory");

        // ---- scores: D = A(x rows 16x32) * B(qt col0), 2 sub-tiles x 4 K-steps ----
        v8f sacc[2];
        #pragma unroll
        for (int sub = 0; sub < 2; ++sub) {
            v8f a = {};
            const __bf16* rowp = &xb[wv][(size_t)(sub * 16 + (lane & 15)) * D_];
            #pragma unroll
            for (int kk = 0; kk < 4; ++kk) {
                v8bf lo = *(const v8bf*)(rowp + kk * 32 + kpart);
                v8bf hi = *(const v8bf*)(rowp + kk * 32 + 16 + kpart);
                v16bf afr = __builtin_shufflevector(lo, hi,
                    0, 1, 2, 3, 4, 5, 6, 7, 8, 9, 10, 11, 12, 13, 14, 15);
                a = __builtin_amdgcn_wmma_f32_16x16x32_bf16(
                        false, afr, false, qB[kk], (short)0, a, false, false);
            }
            sacc[sub] = a;
        }

        // ---- mask + online softmax (scores live in lanes 0 / 16) ----
        float s0[8], s1[8];
        float tmax = FNEG;
        #pragma unroll
        for (int j = 0; j < 8; ++j) {
            float ma0 = madd[wv][sub_off + j];
            float ma1 = madd[wv][16 + sub_off + j];
            s0[j] = (ma0 != 0.f) ? NEGV : sacc[0][j];
            s1[j] = (ma1 != 0.f) ? NEGV : sacc[1][j];
            tmax  = fmaxf(tmax, fmaxf(s0[j], s1[j]));
        }
        if (!ok) tmax = FNEG;
        #pragma unroll
        for (int off = 16; off >= 1; off >>= 1)
            tmax = fmaxf(tmax, __shfl_xor(tmax, off, 32));

        const float m_new = fmaxf(m, tmax);
        const float alpha = __expf(m - m_new);

        float w0[8], w1[8], tsum = 0.f;
        #pragma unroll
        for (int j = 0; j < 8; ++j) {
            w0[j] = __expf(s0[j] - m_new);
            w1[j] = __expf(s1[j] - m_new);
            tsum += w0[j] + w1[j];
        }
        if (!ok) tsum = 0.f;
        #pragma unroll
        for (int off = 16; off >= 1; off >>= 1)
            tsum += __shfl_xor(tsum, off, 32);

        m = m_new;
        l = l * alpha + tsum;

        #pragma unroll
        for (int t = 0; t < 8; ++t)
            #pragma unroll
            for (int j = 0; j < 8; ++j) cacc[t][j] *= alpha;

        // ---- weights as A row 0: the D-layout puts exactly the right rows in lanes 0/16 ----
        v16bf wa = {};
        if (ok) {
            #pragma unroll
            for (int j = 0; j < 8; ++j) {
                wa[j]     = (__bf16)w0[j];
                wa[8 + j] = (__bf16)w1[j];
            }
        }

        // ---- c += w(1x32) * x(32x128): B fragment element s = row 16*(s/8) + 8*(lane/16) + s%8, column lane%16 ----
        const __bf16* myrow = &xb[wv][0];
        #pragma unroll
        for (int dt = 0; dt < 8; ++dt) {
            v16bf bx;
            #pragma unroll
            for (int s = 0; s < 16; ++s)
                bx[s] = myrow[(size_t)(((s >> 3) << 4) + ((lane >> 4) << 3) + (s & 7)) * D_ + dt * 16 + (lane & 15)];
            cacc[dt] = __builtin_amdgcn_wmma_f32_16x16x32_bf16(
                           false, wa, false, bx, (short)0, cacc[dt], false, false);
        }
    }

    // ---- combine 4 wave partials, emit chunk partial (c[128], m, l) ----
    if (lane < 16) {
        #pragma unroll
        for (int dt = 0; dt < 8; ++dt) pc[wv][dt * 16 + lane] = cacc[dt][0];
    }
    if (lane == 0) { pm[wv] = m; pl[wv] = l; }
    __syncthreads();

    const float gm = fmaxf(fmaxf(pm[0], pm[1]), fmaxf(pm[2], pm[3]));
    const float e0 = __expf(pm[0] - gm), e1 = __expf(pm[1] - gm);
    const float e2 = __expf(pm[2] - gm), e3 = __expf(pm[3] - gm);
    const float cf = pc[0][tid] * e0 + pc[1][tid] * e1 + pc[2][tid] * e2 + pc[3][tid] * e3;
    const float lf = pl[0] * e0 + pl[1] * e1 + pl[2] * e2 + pl[3] * e3;

    float* wsp = ws + (size_t)(b * CH + ch) * (D_ + 2);
    wsp[tid] = cf;
    if (tid == 0) { wsp[D_] = gm; wsp[D_ + 1] = lf; }
}

__global__ __launch_bounds__(128) void attn_final_kernel(
    const float* __restrict__ ws, const float* __restrict__ Wv,
    float* __restrict__ out)
{
    __shared__ float cf[D_];
    const int b = blockIdx.x, t = threadIdx.x;
    const float* base = ws + (size_t)b * CH * (D_ + 2);

    float gm = FNEG;
    #pragma unroll
    for (int ch = 0; ch < CH; ++ch) gm = fmaxf(gm, base[ch * (D_ + 2) + D_]);
    float c = 0.f, lf = 0.f;
    #pragma unroll
    for (int ch = 0; ch < CH; ++ch) {
        const float e = __expf(base[ch * (D_ + 2) + D_] - gm);
        c  += base[ch * (D_ + 2) + t] * e;
        lf += base[ch * (D_ + 2) + D_ + 1] * e;
    }
    cf[t] = c;
    __syncthreads();

    const float* wr = Wv + (size_t)t * D_;
    float s = 0.f;
    #pragma unroll 4
    for (int d = 0; d < D_; ++d) s += wr[d] * cf[d];
    out[(size_t)b * D_ + t] = s / lf;
}

extern "C" void kernel_launch(void* const* d_in, const int* in_sizes, int n_in,
                              void* d_out, int out_size, void* d_ws, size_t ws_size,
                              hipStream_t stream) {
    const float* query = (const float*)d_in[0];
    const float* x     = (const float*)d_in[1];
    const int*   amask = (const int*)d_in[2];
    const float* Wq    = (const float*)d_in[3];
    const float* Wk    = (const float*)d_in[4];
    const float* Wv    = (const float*)d_in[5];
    float* out = (float*)d_out;
    float* ws  = (float*)d_ws;   // needs B*CH*(D+2)*4 = ~1.06 MB

    attn_partial_kernel<<<B_ * CH, 128, 0, stream>>>(query, x, amask, Wq, Wk, ws);
    attn_final_kernel<<<B_, 128, 0, stream>>>(ws, Wv, out);
}
